// ModulationBlock_755914244791
// MI455X (gfx1250) — matrix operand order repaired and checked by reading, not compiled, not run on hardware
//
#include <hip/hip_runtime.h>
#include <hip/hip_bf16.h>

// CDNA5 (gfx1250) wave32 WMMA implementation of the ModulationBlock reference.
//
// Pipeline (all f16 WMMA with f32 accumulate):
//   K1 conv_embed : im2col GEMM  M=1024 K=192 N=256 (+bias, LeakyReLU) -> f16
//   K2 qkv        : 3x GEMM      M=1024 K=256 N=256 (+bias); V stored
//                   TRANSPOSED [b][c][p] so the P@V B-fragments are contiguous
//   K3 attn       : per 16-row tile: Q staged to LDS via async-load-to-LDS,
//                   S=QK^T (64KB LDS strip), softmax(/16), top-32 harmonic
//                   reweight scatter, P@V                               -> f32
//   K4 out        : attn_out^T @ w_out^T + b_out, softmax over size-1 axis
//
// Fragment layouts used here (16x16x32 f16, contiguous K order):
//   A (16x32): lane<16 holds row M=lane, K=0..15; lane>=16 holds
//              row M=lane-16, K=16..31  (two contiguous 8-half runs)
//   B (32x16): lane<16 holds col N=lane, K=0..15; lane>=16 col N=lane-16, K=16..31
//   C/D f32  : VGPR v -> row M = v + (lane<16 ? 0 : 8), col N = lane&15

typedef __attribute__((ext_vector_type(16))) _Float16 v16h;
typedef __attribute__((ext_vector_type(8)))  float    v8f;

#define WMMA_F32_F16(A, B, C) \
  __builtin_amdgcn_wmma_f32_16x16x32_f16(false, (A), false, (B), (short)0, (C), false, false)

#if __has_builtin(__builtin_amdgcn_global_load_async_to_lds_b128)
#define HAVE_ASYNC_LDS 1
#define GLOBAL_AS __attribute__((address_space(1)))
#define LDS_AS    __attribute__((address_space(3)))
typedef int v4i __attribute__((vector_size(16)));
#endif

// ---------------------------------------------------------------------------
// K1: patch-embed conv as GEMM.  f[b][p][c] = leaky(sum_k patch[p][k]*W[c][k]+bias)
// block = 256 thr (8 waves); each block does 64 rows x 32 cols (wave -> 16x16 tile)
// ---------------------------------------------------------------------------
__global__ __launch_bounds__(256) void conv_embed_kernel(
    const float* __restrict__ x, const float* __restrict__ w,
    const float* __restrict__ bias, _Float16* __restrict__ f_h)
{
  const int lane = threadIdx.x & 31;
  const int wvid = threadIdx.x >> 5;
  const int b    = blockIdx.z;
  const int rowBase = blockIdx.y * 64 + (wvid >> 1) * 16;
  const int colBase = blockIdx.x * 32 + (wvid & 1) * 16;

  const int m    = lane & 15;
  const int koff = (lane < 16) ? 0 : 16;  // A-fragment K sub-offset
  const int kb   = (lane < 16) ? 0 : 16;  // B-fragment K base
  const int p  = rowBase + m;
  const int py = p >> 5, px = p & 31;
  const float* xb = x + (size_t)b * 3 * 256 * 256 + (size_t)px * 8;
  const int   n   = colBase + m;
  const float* wrow = w + (size_t)n * 192;

  v8f acc = {};
  #pragma unroll
  for (int kc = 0; kc < 6; ++kc) {            // K = 192 = 6 * 32
    const int k0 = kc * 32;
    v16h a;
    {   // run 1: k in [k0+koff, k0+koff+8) -> one contiguous x row segment
      int k = k0 + koff;
      int ci = k >> 6, ky = (k & 63) >> 3;
      const float* src = xb + ((size_t)ci * 256 + py * 8 + ky) * 256;
      #pragma unroll
      for (int e = 0; e < 8; ++e) a[e] = (_Float16)src[e];
    }
    {   // run 2: k in [k0+8+koff, ...+8)
      int k = k0 + 8 + koff;
      int ci = k >> 6, ky = (k & 63) >> 3;
      const float* src = xb + ((size_t)ci * 256 + py * 8 + ky) * 256;
      #pragma unroll
      for (int e = 0; e < 8; ++e) a[e + 8] = (_Float16)src[e];
    }
    v16h bf;
    const float* src = wrow + k0 + kb;        // B[k][n] = conv_w[n][k], K-contiguous
    #pragma unroll
    for (int e = 0; e < 16; ++e) bf[e] = (_Float16)src[e];
    acc = WMMA_F32_F16(a, bf, acc);
  }
  #pragma unroll
  for (int v = 0; v < 8; ++v) {
    int row = rowBase + v + ((lane < 16) ? 0 : 8);
    int col = colBase + (lane & 15);
    float val = acc[v] + bias[col];
    val = (val > 0.0f) ? val : 0.01f * val;   // LeakyReLU(0.01)
    f_h[((size_t)b * 1024 + row) * 256 + col] = (_Float16)val;
  }
}

// ---------------------------------------------------------------------------
// K2: q/k/v = f @ W^T + b.   grid.z = b*3 + which
//   q,k stored [b][p][c]; v stored transposed [b][c][p] for the P@V B-frags.
// ---------------------------------------------------------------------------
__global__ __launch_bounds__(256) void qkv_kernel(
    const _Float16* __restrict__ f_h,
    const float* __restrict__ wq, const float* __restrict__ bq,
    const float* __restrict__ wk, const float* __restrict__ bk,
    const float* __restrict__ wv, const float* __restrict__ bv,
    _Float16* __restrict__ q_h, _Float16* __restrict__ k_h,
    _Float16* __restrict__ v_t)
{
  const int lane = threadIdx.x & 31;
  const int wvid = threadIdx.x >> 5;
  const int z = blockIdx.z;
  const int b = z / 3, which = z % 3;
  const float* W  = (which == 0) ? wq : (which == 1) ? wk : wv;
  const float* Bv = (which == 0) ? bq : (which == 1) ? bk : bv;

  const int rowBase = blockIdx.y * 64 + (wvid >> 1) * 16;
  const int colBase = blockIdx.x * 32 + (wvid & 1) * 16;
  const int m    = lane & 15;
  const int koff = (lane < 16) ? 0 : 16;
  const int kb   = (lane < 16) ? 0 : 16;
  const _Float16* arow = f_h + ((size_t)b * 1024 + rowBase + m) * 256;
  const float*    brow = W + (size_t)(colBase + m) * 256;   // B[k][n]=W[n][k]

  v8f acc = {};
  #pragma unroll
  for (int kc = 0; kc < 8; ++kc) {            // K = 256 = 8 * 32
    const int k0 = kc * 32;
    __builtin_prefetch(arow + k0 + 64, 0, 0); // -> global_prefetch_b8
    v16h a;
    #pragma unroll
    for (int e = 0; e < 8; ++e) {
      a[e]     = arow[k0 + koff + e];
      a[e + 8] = arow[k0 + 8 + koff + e];
    }
    v16h bf;
    #pragma unroll
    for (int e = 0; e < 16; ++e) bf[e] = (_Float16)brow[k0 + kb + e];
    acc = WMMA_F32_F16(a, bf, acc);
  }
  if (which == 2) {                            // V: transposed store [b][c][p]
    #pragma unroll
    for (int v = 0; v < 8; ++v) {
      int row = rowBase + v + ((lane < 16) ? 0 : 8);
      int col = colBase + (lane & 15);
      v_t[((size_t)b * 256 + col) * 1024 + row] = (_Float16)(acc[v] + Bv[col]);
    }
  } else {
    _Float16* out = (which == 0) ? q_h : k_h;
    #pragma unroll
    for (int v = 0; v < 8; ++v) {
      int row = rowBase + v + ((lane < 16) ? 0 : 8);
      int col = colBase + (lane & 15);
      out[((size_t)b * 1024 + row) * 256 + col] = (_Float16)(acc[v] + Bv[col]);
    }
  }
}

// ---------------------------------------------------------------------------
// K3: fused attention for one (batch, 16-row) tile.
//   Q tile async-staged to LDS (aliases the S strip; consumed into registers
//   before S is written).  S strip: 16 x 1024 f32 = 64KB.
// ---------------------------------------------------------------------------
__global__ __launch_bounds__(256) void attn_kernel(
    const _Float16* __restrict__ q_h, const _Float16* __restrict__ k_h,
    const _Float16* __restrict__ v_t, const int* __restrict__ simi,
    float* __restrict__ attn_out)
{
  __shared__ union {
    float    S[16][1024];                     // 64 KB score strip
    _Float16 Q[16][256];                      // 8 KB staged Q tile (phase 0 only)
  } sh;
  const int lane = threadIdx.x & 31;
  const int wvid = threadIdx.x >> 5;
  const int b = blockIdx.y;
  const int rowBase = blockIdx.x * 16;
  const size_t bOff = (size_t)b * 1024 * 256;

  const int m    = lane & 15;
  const int koff = (lane < 16) ? 0 : 16;
  const int kb   = (lane < 16) ? 0 : 16;

  // ---- Phase 0a: stage Q tile (16x256 f16 = 8KB) into LDS once ----
  {
    const _Float16* gq = q_h + bOff + (size_t)rowBase * 256;
#ifdef HAVE_ASYNC_LDS
    #pragma unroll
    for (int i = 0; i < 2; ++i) {             // 512 x b128 chunks, 2 per thread
      const int c = threadIdx.x + i * 256;
      __builtin_amdgcn_global_load_async_to_lds_b128(
          (GLOBAL_AS v4i*)(gq + c * 8),
          (LDS_AS v4i*)(&sh.Q[0][0] + c * 8), 0, 0);
    }
    asm volatile("s_wait_asynccnt 0" ::: "memory");
    __syncthreads();
#else
    for (int i = threadIdx.x; i < 16 * 256; i += 256) (&sh.Q[0][0])[i] = gq[i];
    __syncthreads();
#endif
  }

  // ---- Phase 0b: Q fragments for this tile (held in 64 VGPRs) ----
  v16h qf[8];
  {
    const _Float16* qs = &sh.Q[m][0];
    #pragma unroll
    for (int kc = 0; kc < 8; ++kc) {
      #pragma unroll
      for (int e = 0; e < 8; ++e) {
        qf[kc][e]     = qs[kc * 32 + koff + e];
        qf[kc][e + 8] = qs[kc * 32 + 8 + koff + e];
      }
    }
  }
  __syncthreads();                            // Q region is about to alias S

  // ---- Phase 1: S = Q @ K^T (raw scores); 8 waves x 8 column tiles ----
  for (int ctl = 0; ctl < 8; ++ctl) {
    const int colBase = (wvid + ctl * 8) * 16;
    const _Float16* krow = k_h + bOff + (size_t)(colBase + m) * 256 + kb;
    v8f acc = {};
    #pragma unroll
    for (int kc = 0; kc < 8; ++kc) {
      __builtin_prefetch(krow + kc * 32 + 64, 0, 0);
      v16h bf;                                // B[k][j] = K[j][k], K-contiguous
      #pragma unroll
      for (int e = 0; e < 16; ++e) bf[e] = krow[kc * 32 + e];
      acc = WMMA_F32_F16(qf[kc], bf, acc);
    }
    #pragma unroll
    for (int v = 0; v < 8; ++v)
      sh.S[v + ((lane < 16) ? 0 : 8)][colBase + m] = acc[v];
  }
  __syncthreads();

  // ---- Phase 2: row softmax of S/16; wave w owns rows {2w, 2w+1} ----
  for (int rr = 0; rr < 2; ++rr) {
    const int r = wvid * 2 + rr;
    float mx = -3.4e38f;
    for (int col = lane; col < 1024; col += 32)
      mx = fmaxf(mx, sh.S[r][col] * 0.0625f);
    #pragma unroll
    for (int s = 16; s > 0; s >>= 1) mx = fmaxf(mx, __shfl_xor(mx, s, 32));
    float sum = 0.0f;
    for (int col = lane; col < 1024; col += 32)
      sum += __expf(sh.S[r][col] * 0.0625f - mx);
    #pragma unroll
    for (int s = 16; s > 0; s >>= 1) sum += __shfl_xor(sum, s, 32);
    const float inv = 1.0f / sum;
    for (int col = lane; col < 1024; col += 32)
      sh.S[r][col] = __expf(sh.S[r][col] * 0.0625f - mx) * inv;
  }
  __syncthreads();

  // ---- Phase 3: top-32 harmonic reweight (gather all, then ordered writes) ----
  if (threadIdx.x < 16) {
    const int r = threadIdx.x;
    const int grow = rowBase + r;
    const int* ip = simi + ((size_t)b * 1024 + grow) * 32;
    float H = 0.0f;
    #pragma unroll
    for (int kk = 0; kk < 32; ++kk) H += 1.0f / (float)(kk + 1);
    const float invH = 1.0f / H;
    float vals[32]; int cols[32];
    #pragma unroll
    for (int kk = 0; kk < 32; ++kk) {
      int j = ip[kk];
      if (j < 0) j = grow;                    // -1 sentinel -> diagonal
      cols[kk] = j;
      vals[kk] = sh.S[r][j];                  // take_along_axis before set
    }
    #pragma unroll
    for (int kk = 0; kk < 32; ++kk)           // in-order: last write wins
      sh.S[r][cols[kk]] = vals[kk] * (1.0f / (float)(kk + 1)) * invH;
  }
  __syncthreads();

  // ---- Phase 4: attn_out = P @ V; wave w owns output cols [32w, 32w+32) ----
  {
    const int c0 = wvid * 32;
    // V transposed: B[k][c] = v_t[b][c][k] -> per-lane K-contiguous runs
    const _Float16* vr0 = v_t + ((size_t)b * 256 + c0 + m) * 1024 + kb;
    const _Float16* vr1 = v_t + ((size_t)b * 256 + c0 + 16 + m) * 1024 + kb;
    v8f acc0 = {}, acc1 = {};
    for (int kc = 0; kc < 32; ++kc) {         // K = 1024 = 32 * 32
      const int k0 = kc * 32;
      __builtin_prefetch(vr0 + k0 + 64, 0, 0);
      v16h a;                                 // f32 LDS -> f16 A-frag
      #pragma unroll
      for (int e = 0; e < 8; ++e) {
        a[e]     = (_Float16)sh.S[m][k0 + koff + e];
        a[e + 8] = (_Float16)sh.S[m][k0 + 8 + koff + e];
      }
      v16h b0, b1;
      #pragma unroll
      for (int e = 0; e < 16; ++e) {
        b0[e] = vr0[k0 + e];
        b1[e] = vr1[k0 + e];
      }
      acc0 = WMMA_F32_F16(a, b0, acc0);
      acc1 = WMMA_F32_F16(a, b1, acc1);
    }
    #pragma unroll
    for (int v = 0; v < 8; ++v) {
      const int row = rowBase + v + ((lane < 16) ? 0 : 8);
      attn_out[bOff + (size_t)row * 256 + c0 + m]      = acc0[v];
      attn_out[bOff + (size_t)row * 256 + c0 + 16 + m] = acc1[v];
    }
  }
}

// ---------------------------------------------------------------------------
// K4: out[b][c] = sum_p attn_out[b][p][c] * w_out[p] + b_out; softmax(size-1)
// ---------------------------------------------------------------------------
__global__ __launch_bounds__(256) void out_kernel(
    const float* __restrict__ attn_out, const float* __restrict__ w_out,
    const float* __restrict__ b_out, float* __restrict__ out)
{
  const int b = blockIdx.x;
  const int c = threadIdx.x;
  float acc = b_out[0];
  const float* ap = attn_out + (size_t)b * 1024 * 256 + c;
  for (int p = 0; p < 1024; ++p)
    acc += ap[(size_t)p * 256] * w_out[p];
  // softmax over the trailing axis of size 1 (kept faithful incl. NaN prop)
  float e = expf(acc - acc);
  out[b * 256 + c] = e / e;
}

// ---------------------------------------------------------------------------
extern "C" void kernel_launch(void* const* d_in, const int* in_sizes, int n_in,
                              void* d_out, int out_size, void* d_ws, size_t ws_size,
                              hipStream_t stream)
{
  const float* x      = (const float*)d_in[0];
  const float* conv_w = (const float*)d_in[1];
  const float* conv_b = (const float*)d_in[2];
  const float* wq     = (const float*)d_in[3];
  const float* bq     = (const float*)d_in[4];
  const float* wk     = (const float*)d_in[5];
  const float* bk     = (const float*)d_in[6];
  const float* wv     = (const float*)d_in[7];
  const float* bv     = (const float*)d_in[8];
  const float* w_out  = (const float*)d_in[9];
  const float* b_out  = (const float*)d_in[10];
  const int*   simi   = (const int*)d_in[11];
  float* out = (float*)d_out;

  // Workspace: f16 f/q/k/v_t (4 x 16MB) + f32 attn_out (32MB) ~= 100MB
  char* ws = (char*)d_ws;
  const size_t FH = (size_t)32 * 1024 * 256;
  _Float16* f_h = (_Float16*)ws;  ws += FH * sizeof(_Float16);
  _Float16* q_h = (_Float16*)ws;  ws += FH * sizeof(_Float16);
  _Float16* k_h = (_Float16*)ws;  ws += FH * sizeof(_Float16);
  _Float16* v_t = (_Float16*)ws;  ws += FH * sizeof(_Float16);
  float* attn_o = (float*)ws;

  conv_embed_kernel<<<dim3(8, 16, 32), 256, 0, stream>>>(x, conv_w, conv_b, f_h);
  qkv_kernel<<<dim3(8, 16, 96), 256, 0, stream>>>(f_h, wq, bq, wk, bk, wv, bv,
                                                  q_h, k_h, v_t);
  attn_kernel<<<dim3(64, 32, 1), 256, 0, stream>>>(q_h, k_h, v_t, simi, attn_o);
  out_kernel<<<dim3(32, 1, 1), 256, 0, stream>>>(attn_o, w_out, b_out, out);
}
